// GMABSE3_32813550141527
// MI455X (gfx1250) — matrix operand order repaired and checked by reading, not compiled, not run on hardware
//
#include <hip/hip_runtime.h>
#include <stdint.h>

typedef __attribute__((ext_vector_type(16))) _Float16 v16h;
typedef __attribute__((ext_vector_type(8)))  float    v8f;

#define N_HEADS 8
#define INV_SQRT_D 0.08838834764831845f  /* 1/sqrt(128) */

// Monotonic order-preserving float <-> u32 mapping (for atomic max on floats).
__device__ __forceinline__ unsigned f32_ord(float f) {
  unsigned u = __float_as_uint(f);
  return (u & 0x80000000u) ? ~u : (u | 0x80000000u);
}
__device__ __forceinline__ float ord_f32(unsigned u) {
  return __uint_as_float((u & 0x80000000u) ? (u & 0x7FFFFFFFu) : ~u);
}

// ---------------------------------------------------------------------------
// Init: zero d_out, init segment-max (encoded -inf == 0) and segment-sum.
// ---------------------------------------------------------------------------
__global__ void init_kernel(float* __restrict__ out, int out_size,
                            unsigned* __restrict__ m_u, float* __restrict__ s,
                            int nh) {
  int i = blockIdx.x * blockDim.x + threadIdx.x;
  if (i < out_size) out[i] = 0.0f;
  if (i < nh) { m_u[i] = 0u; s[i] = 0.0f; }
}

// ---------------------------------------------------------------------------
// Kernel 1: per-edge, per-head QK dot via v_wmma_f32_16x16x32_f16.
// One wave handles a tile of 16 edges; 8 WMMAs (one per head).
//   A (16x32 f16): row m = key of edge (ebase+m), dims 0..15, K 16..31 = 0.
//     lanes 0-15  : a[0..15] = dims 0..15 (K0..15, same order as b)
//     lanes 16-31 : 0 (K16..31)
//   B (32x16 f16): col n = query of dst(edge ebase+n).
//     lanes 0-15  : b[0..15] = dims 0..15 ; lanes 16-31 : 0 (K16..31)
//   Diagonal D[n][n]: lane n (n<8) at c[n]; lane n+16 (n>=8) at c[n-8].
//   Extraction = one-hot mask dot (8 FMAs), results buffered per lane and
//   written as two coalesced float4 stores per owned edge.
// ---------------------------------------------------------------------------
__global__ void qk_dot_wmma(const float* __restrict__ q0, const float* __restrict__ q1,
                            const float* __restrict__ k0, const float* __restrict__ k1,
                            const int* __restrict__ edge_dst,
                            float* __restrict__ e_out, int n_edges) {
  const int lane = threadIdx.x & 31;
  const int wave = threadIdx.x >> 5;
  const int tile = blockIdx.x * (blockDim.x >> 5) + wave;
  const int ntiles = (n_edges + 15) >> 4;
  if (tile >= ntiles) return;                 // wave-uniform: EXEC stays all-1s
  const int ebase = tile << 4;
  int eidx = ebase + (lane & 15);
  if (eidx >= n_edges) eidx = n_edges - 1;    // clamp keeps all lanes active
  const int dst = edge_dst[eidx];

  const float* kk0 = k0 + (size_t)eidx * 32;  // [32] degree-0: 4 dims/head
  const float* kk1 = k1 + (size_t)eidx * 96;  // [96] degree-1: 12 dims/head
  const float* qq0 = q0 + (size_t)dst * 32;
  const float* qq1 = q1 + (size_t)dst * 96;

  // Which diagonal VGPR does this lane own? lanes 0-7 -> edge=lane, r=lane;
  // lanes 24-31 -> edge=lane-16, r=lane-24; other lanes own nothing.
  const int owner = (lane < 8) ? lane : ((lane >= 24) ? (lane - 16) : -1);
  const int ridx  = (lane < 8) ? lane : (lane - 24);
  float msk[8];
  #pragma unroll
  for (int r = 0; r < 8; ++r)
    msk[r] = (owner >= 0 && ridx == r) ? INV_SQRT_D : 0.0f;

  float res[N_HEADS];

  #pragma unroll
  for (int h = 0; h < N_HEADS; ++h) {
    v16h a = {};
    v16h b = {};
    if (lane < 16) {
      float4 ka = *(const float4*)(kk0 + h * 4);       // k dims 0..3
      float4 kb = *(const float4*)(kk1 + h * 12);      // k dims 4..7
      a[0] = (_Float16)ka.x; a[1] = (_Float16)ka.y;
      a[2] = (_Float16)ka.z; a[3] = (_Float16)ka.w;
      a[4] = (_Float16)kb.x; a[5] = (_Float16)kb.y;
      a[6] = (_Float16)kb.z; a[7] = (_Float16)kb.w;
      float4 qa = *(const float4*)(qq0 + h * 4);       // q dims 0..3
      float4 qb = *(const float4*)(qq1 + h * 12);      // q dims 4..7
      float4 qc = *(const float4*)(qq1 + h * 12 + 4);  // q dims 8..11
      float4 qd = *(const float4*)(qq1 + h * 12 + 8);  // q dims 12..15
      b[0]  = (_Float16)qa.x; b[1]  = (_Float16)qa.y;
      b[2]  = (_Float16)qa.z; b[3]  = (_Float16)qa.w;
      b[4]  = (_Float16)qb.x; b[5]  = (_Float16)qb.y;
      b[6]  = (_Float16)qb.z; b[7]  = (_Float16)qb.w;
      b[8]  = (_Float16)qc.x; b[9]  = (_Float16)qc.y;
      b[10] = (_Float16)qc.z; b[11] = (_Float16)qc.w;
      b[12] = (_Float16)qd.x; b[13] = (_Float16)qd.y;
      b[14] = (_Float16)qd.z; b[15] = (_Float16)qd.w;
      float4 kc = *(const float4*)(kk1 + h * 12 + 4);  // k dims 8..11
      float4 kd = *(const float4*)(kk1 + h * 12 + 8);  // k dims 12..15
      a[8] = (_Float16)kc.x; a[9] = (_Float16)kc.y;
      a[10] = (_Float16)kc.z; a[11] = (_Float16)kc.w;
      a[12] = (_Float16)kd.x; a[13] = (_Float16)kd.y;
      a[14] = (_Float16)kd.z; a[15] = (_Float16)kd.w;
    }
    v8f c = {};
    c = __builtin_amdgcn_wmma_f32_16x16x32_f16(
        /*neg_a=*/false, a, /*neg_b=*/false, b,
        /*c_mod=*/(short)0, c, /*reuse_a=*/false, /*reuse_b=*/false);

    // One-hot masked dot picks this lane's diagonal element (scale folded in).
    float val = 0.0f;
    #pragma unroll
    for (int r = 0; r < 8; ++r) val = fmaf(c[r], msk[r], val);
    res[h] = val;
  }

  // Coalesced row store: e_out[eo, 0..7] as two b128 stores per owned edge.
  if (owner >= 0) {
    int eo = ebase + owner;
    if (eo < n_edges) {
      float4* p = (float4*)(e_out + (size_t)eo * N_HEADS);
      p[0] = make_float4(res[0], res[1], res[2], res[3]);
      p[1] = make_float4(res[4], res[5], res[6], res[7]);
    }
  }
}

// ---------------------------------------------------------------------------
// Kernel 2: segment max over dst nodes (encoded-u32 atomic max).
// ---------------------------------------------------------------------------
__global__ void seg_max(const float* __restrict__ e, const int* __restrict__ edge_dst,
                        unsigned* __restrict__ m_u, int total) {
  int idx = blockIdx.x * blockDim.x + threadIdx.x;
  if (idx >= total) return;
  int ed = idx >> 3, h = idx & 7;
  atomicMax(&m_u[(size_t)edge_dst[ed] * N_HEADS + h], f32_ord(e[idx]));
}

// ---------------------------------------------------------------------------
// Kernel 3: ex = exp(e - m[dst]); in-place overwrite; segment sum via f32 atomics.
// ---------------------------------------------------------------------------
__global__ void seg_expsum(float* __restrict__ e, const int* __restrict__ edge_dst,
                           const unsigned* __restrict__ m_u, float* __restrict__ s,
                           int total) {
  int idx = blockIdx.x * blockDim.x + threadIdx.x;
  if (idx >= total) return;
  int ed = idx >> 3, h = idx & 7;
  size_t seg = (size_t)edge_dst[ed] * N_HEADS + h;
  float m = ord_f32(m_u[seg]);
  float ex = __expf(e[idx] - m);
  e[idx] = ex;
  atomicAdd(&s[seg], ex);
}

// ---------------------------------------------------------------------------
// Kernel 4: a = ex/s[dst]; out[dst] += a*v  (16 floats per (edge,head)).
// d_out layout: out0 [N,32,1] followed by out1 [N,32,3], flat.
// ---------------------------------------------------------------------------
__global__ void seg_agg(const float* __restrict__ exv, const int* __restrict__ edge_dst,
                        const float* __restrict__ s,
                        const float* __restrict__ v0, const float* __restrict__ v1,
                        float* __restrict__ out, int n_nodes, int total) {
  int idx = blockIdx.x * blockDim.x + threadIdx.x;
  if (idx >= total) return;
  int ed = idx >> 3, h = idx & 7;
  int dst = edge_dst[ed];
  float a = exv[idx] / s[(size_t)dst * N_HEADS + h];
  float4 w0 = *(const float4*)(v0 + (size_t)ed * 32 + h * 4);
  float4 w1 = *(const float4*)(v1 + (size_t)ed * 96 + h * 12);
  float4 w2 = *(const float4*)(v1 + (size_t)ed * 96 + h * 12 + 4);
  float4 w3 = *(const float4*)(v1 + (size_t)ed * 96 + h * 12 + 8);
  float* o0 = out + (size_t)dst * 32 + h * 4;
  float* o1 = out + (size_t)n_nodes * 32 + (size_t)dst * 96 + h * 12;
  atomicAdd(&o0[0], a * w0.x); atomicAdd(&o0[1], a * w0.y);
  atomicAdd(&o0[2], a * w0.z); atomicAdd(&o0[3], a * w0.w);
  atomicAdd(&o1[0], a * w1.x); atomicAdd(&o1[1],  a * w1.y);
  atomicAdd(&o1[2], a * w1.z); atomicAdd(&o1[3],  a * w1.w);
  atomicAdd(&o1[4], a * w2.x); atomicAdd(&o1[5],  a * w2.y);
  atomicAdd(&o1[6], a * w2.z); atomicAdd(&o1[7],  a * w2.w);
  atomicAdd(&o1[8], a * w3.x); atomicAdd(&o1[9],  a * w3.y);
  atomicAdd(&o1[10], a * w3.z); atomicAdd(&o1[11], a * w3.w);
}

// ---------------------------------------------------------------------------
extern "C" void kernel_launch(void* const* d_in, const int* in_sizes, int n_in,
                              void* d_out, int out_size, void* d_ws, size_t ws_size,
                              hipStream_t stream) {
  const float* q0 = (const float*)d_in[0];
  const float* q1 = (const float*)d_in[1];
  const float* k0 = (const float*)d_in[2];
  const float* k1 = (const float*)d_in[3];
  const float* v0 = (const float*)d_in[4];
  const float* v1 = (const float*)d_in[5];
  const int*   edge_dst = (const int*)d_in[6];

  const int n_nodes = in_sizes[0] / 32;   // q0 = [N,32,1]
  const int n_edges = in_sizes[2] / 32;   // k0 = [E,32,1]
  const int EH = n_edges * N_HEADS;
  const int NH = n_nodes * N_HEADS;

  // Workspace: e/ex [E,H] f32 | m_u [N,H] u32 | s [N,H] f32  (~21.8 MB)
  float*    e_buf = (float*)d_ws;
  unsigned* m_u   = (unsigned*)(e_buf + (size_t)EH);
  float*    s_buf = (float*)(m_u + (size_t)NH);
  float*    out   = (float*)d_out;

  {
    int n = out_size > NH ? out_size : NH;
    init_kernel<<<(n + 255) / 256, 256, 0, stream>>>(out, out_size, m_u, s_buf, NH);
  }
  {
    const int ntiles = (n_edges + 15) / 16;    // one wave per 16-edge tile
    const int wpb = 8;                         // 8 waves (256 threads) / block
    const int blocks = (ntiles + wpb - 1) / wpb;
    qk_dot_wmma<<<blocks, wpb * 32, 0, stream>>>(q0, q1, k0, k1, edge_dst,
                                                 e_buf, n_edges);
  }
  seg_max   <<<(EH + 255) / 256, 256, 0, stream>>>(e_buf, edge_dst, m_u, EH);
  seg_expsum<<<(EH + 255) / 256, 256, 0, stream>>>(e_buf, edge_dst, m_u, s_buf, EH);
  seg_agg   <<<(EH + 255) / 256, 256, 0, stream>>>(e_buf, edge_dst, s_buf, v0, v1,
                                                   out, n_nodes, EH);
}
